// VarMaskedFastLSTM_30494267802052
// MI455X (gfx1250) — matrix operand order repaired and checked by reading, not compiled, not run on hardware
//
#include <hip/hip_runtime.h>

typedef __bf16 bf16_t;
typedef __attribute__((ext_vector_type(16))) __bf16 v16bf;
typedef __attribute__((ext_vector_type(8)))  __bf16 v8bf;
typedef __attribute__((ext_vector_type(4)))  __bf16 v4bf;
typedef __attribute__((ext_vector_type(8)))  float   v8f;
typedef __attribute__((ext_vector_type(4)))  float   v4f;
typedef int v4i_vs __attribute__((vector_size(16)));   // matches async-LDS builtin param

#define T_STEPS 256
#define BATCH   64
#define HID     512
#define NBLK    16          // blocks in the persistent scan kernel (8 hTiles x 2 dirs)

#if defined(__AMDGCN__) && \
    __has_builtin(__builtin_amdgcn_global_load_async_to_lds_b128) && \
    __has_builtin(__builtin_amdgcn_s_wait_asynccnt)
#define HAVE_ASYNC_LDS 1
#endif

// ---------------------------------------------------------------------------
// Transpose fp32 weights [DG][K][N] -> bf16 [DG][N][K]
// ---------------------------------------------------------------------------
__global__ void wt_kernel(const float* __restrict__ src, bf16_t* __restrict__ dst,
                          int K, int N, long total) {
    long i = (long)blockIdx.x * blockDim.x + threadIdx.x;
    if (i >= total) return;
    long kn = (long)K * N;
    long dg = i / kn;
    long rem = i - dg * kn;
    long k = rem / N;
    long n = rem - k * N;
    dst[(dg * N + n) * K + k] = (bf16_t)src[i];
}

__global__ void zero_kernel(float* __restrict__ p, int n) {
    int i = blockIdx.x * blockDim.x + threadIdx.x;
    if (i < n) p[i] = 0.0f;
}

// ---------------------------------------------------------------------------
// gx GEMM: gx[d][t][g][h][b] = (X[t,b,:] * noise[d,g,b,:]) @ W_ih[d,g] + b_ih + b_hh
// Block: (mTile, nTile, d). 512 threads = 16 waves. One block does ALL 4 gates,
// so the x-tile is read from HBM once and expanded to 4 bf16 LDS tiles.
// ---------------------------------------------------------------------------
__global__ __launch_bounds__(512) void gx_gemm_kernel(
    const float*  __restrict__ X,      // [M][K]
    const float*  __restrict__ noise,  // [8][B][K]
    const bf16_t* __restrict__ wT,     // [8][512][K]   (N-major)
    const float*  __restrict__ b_ih,   // [8][512]
    const float*  __restrict__ b_hh,   // [8][512]
    float*        __restrict__ gx,     // [2][T][4][512][64]
    int K)
{
    __shared__ bf16_t lsA[4][128 * 40];   // per-gate A tiles, 128 rows x 32 k

    const int tid  = threadIdx.x;
    const int lane = tid & 31;
    const int wave = tid >> 5;            // 0..15
    const int g    = wave >> 2;
    const int mw   = wave & 3;
    const int mBase = blockIdx.x * 128;
    const int nBase = blockIdx.y * 64;
    const int d     = blockIdx.z;

    v8f acc[2][4] = {};

    const bf16_t* wDG = wT + (long)(d * 4 + g) * 512 * K;

    const int srow = tid >> 2;
    const int skc  = (tid & 3) * 8;
    const int sm   = mBase + srow;
    const int sb   = sm & (BATCH - 1);

    for (int kb = 0; kb < K; kb += 32) {
        __syncthreads();
        {
            const float* xp = X + (long)sm * K + kb + skc;
            if (kb + 32 < K) __builtin_prefetch(xp + 32, 0, 1);
            v4f x0 = *(const v4f*)xp, x1 = *(const v4f*)(xp + 4);
#pragma unroll
            for (int gg = 0; gg < 4; ++gg) {
                const float* np = noise + ((long)(d * 4 + gg) * BATCH + sb) * K + kb + skc;
                v4f n0 = *(const v4f*)np, n1 = *(const v4f*)(np + 4);
                v8bf a;
#pragma unroll
                for (int j = 0; j < 4; ++j) {
                    a[j]     = (bf16_t)(x0[j] * n0[j]);
                    a[4 + j] = (bf16_t)(x1[j] * n1[j]);
                }
                *(v8bf*)&lsA[gg][srow * 40 + skc] = a;
            }
        }
        __syncthreads();

        v16bf bf[4];
#pragma unroll
        for (int j = 0; j < 4; ++j) {
            int col = nBase + j * 16 + (lane & 15);
            int k0  = kb + (lane >> 4) * 16;
            const bf16_t* p = wDG + (long)col * K + k0;
            ((v8bf*)&bf[j])[0] = *(const v8bf*)p;
            ((v8bf*)&bf[j])[1] = *(const v8bf*)(p + 8);
        }
        v16bf af[2];
#pragma unroll
        for (int i = 0; i < 2; ++i) {
            int row = mw * 32 + i * 16 + (lane & 15);
            int k0  = (lane >> 4) * 16;
            const bf16_t* p = &lsA[g][row * 40 + k0];
            ((v8bf*)&af[i])[0] = *(const v8bf*)p;
            ((v8bf*)&af[i])[1] = *(const v8bf*)(p + 8);
        }
#pragma unroll
        for (int i = 0; i < 2; ++i)
#pragma unroll
            for (int j = 0; j < 4; ++j)
                acc[i][j] = __builtin_amdgcn_wmma_f32_16x16x32_bf16(
                    false, af[i], false, bf[j], (short)0, acc[i][j], false, false);
    }

    // Epilogue: acc element r = batch row; contiguous in [b] axis -> v8f store
#pragma unroll
    for (int i = 0; i < 2; ++i) {
        int rbase = mw * 32 + i * 16 + ((lane & 16) >> 1);
        int m0 = mBase + rbase;
        int t  = m0 >> 6;
        int b0 = m0 & 63;
#pragma unroll
        for (int j = 0; j < 4; ++j) {
            int col = nBase + j * 16 + (lane & 15);
            float bias = b_ih[(d * 4 + g) * 512 + col] + b_hh[(d * 4 + g) * 512 + col];
            v8f v = acc[i][j];
#pragma unroll
            for (int r = 0; r < 8; ++r) v[r] += bias;
            long off = ((((long)d * T_STEPS + t) * 4 + g) * 512 + col) * 64 + b0;
            *(v8f*)&gx[off] = v;
        }
    }
}

// ---------------------------------------------------------------------------
// Persistent recurrent scan. Grid: (hTile 0..7, dir 0..1) = 16 blocks.
// The block's whole w_hh slice (4 gates x 64 cols x 512 K, bf16 = 256KB) is
// loaded ONCE into LDS (async-to-LDS when available) with an XOR bank swizzle;
// hA staging and the gate-exchange buffer share the remaining 64KB (disjoint
// live ranges). Total LDS = 320KB (one workgroup per WGP).
// ---------------------------------------------------------------------------
__global__ __launch_bounds__(512) void lstm_scan_kernel(
    const float*  __restrict__ gx,       // [2][T][4][512][64]
    const float*  __restrict__ mask,     // [T][B]
    const float*  __restrict__ noise_h,  // [8][B][512]
    const bf16_t* __restrict__ wT,       // [8][512][512]
    float* __restrict__ stateH0, float* __restrict__ stateC0,  // [2][B][512]
    float* __restrict__ stateH1, float* __restrict__ stateC1,
    float*        __restrict__ outp,     // [T][B][2H]
    unsigned*     __restrict__ barrier_ctr)
{
    // 4 gates x 64 cols x 64 chunks x 8 bf16; physical chunk = logical ^ (col&15)
    __shared__ bf16_t wLds[4 * 64 * 64 * 8];          // 256 KB
    __shared__ __align__(16) char shScratch[65536];   // hA (20KB) / gBuf (64KB) union
    bf16_t* hA   = (bf16_t*)shScratch;                // [4][64*40]
    float*  gBuf = (float*)shScratch;                 // [4][4096]

    const int tid  = threadIdx.x;
    const int lane = tid & 31;
    const int wave = tid >> 5;           // 0..15
    const int g    = wave >> 2;
    const int nSub = wave & 3;
    const int hTile = blockIdx.x;        // 0..7
    const int d     = blockIdx.y;        // 0..1
    const long sBase = (long)d * BATCH * HID;

    // ---- one-time weight fill: 16384 chunks of 16B, 32 per thread ----
    for (int cc = tid; cc < 16384; cc += 512) {
        int fg    = cc >> 12;
        int colL  = (cc >> 6) & 63;
        int clog  = cc & 63;
        const bf16_t* src = wT + ((long)(d * 4 + fg) * 512 + hTile * 64 + colL) * 512 + clog * 8;
        bf16_t* dst = &wLds[(((fg * 64 + colL) << 6) + (clog ^ (colL & 15))) * 8];
#if defined(HAVE_ASYNC_LDS)
        __builtin_amdgcn_global_load_async_to_lds_b128(
            (v4i_vs*)src, (v4i_vs*)dst, 0, 0);
#else
        *(v8bf*)dst = *(const v8bf*)src;
#endif
    }
#if defined(HAVE_ASYNC_LDS)
    __builtin_amdgcn_s_wait_asynccnt(0);
#endif
    __syncthreads();

    for (int s = 0; s < T_STEPS; ++s) {
        const int ti = d ? (T_STEPS - 1 - s) : s;
        const float* hIn  = ((s & 1) ? stateH1 : stateH0) + sBase;
        const float* cIn  = ((s & 1) ? stateC1 : stateC0) + sBase;
        float*       hOut = ((s & 1) ? stateH0 : stateH1) + sBase;
        float*       cOut = ((s & 1) ? stateC0 : stateC1) + sBase;

        // Prefetch next step's gx slice for this block
        if (s + 1 < T_STEPS) {
            int tn = d ? (T_STEPS - 2 - s) : (s + 1);
            const char* pb = (const char*)(gx +
                ((((long)d * T_STEPS + tn) * 4 + (tid >> 7)) * 512 + hTile * 64) * 64);
            __builtin_prefetch(pb + (long)(tid & 127) * 128, 0, 1);
        }

        v8f acc[4] = {};

        for (int kb = 0; kb < HID; kb += 32) {
            __syncthreads();
            {
                int row = tid >> 3;          // 0..63
                int kc  = (tid & 7) * 4;     // 0..28
                const float* hp = hIn + (long)row * HID + kb + kc;
                v4f hv = *(const v4f*)hp;
#pragma unroll
                for (int gg = 0; gg < 4; ++gg) {
                    const float* np = noise_h + ((long)(d * 4 + gg) * BATCH + row) * HID + kb + kc;
                    v4f nv = *(const v4f*)np;
                    v4bf a;
#pragma unroll
                    for (int j = 0; j < 4; ++j) a[j] = (bf16_t)(hv[j] * nv[j]);
                    *(v4bf*)&hA[gg * (64 * 40) + row * 40 + kc] = a;
                }
            }
            __syncthreads();

            // B fragment from LDS weight cache (xor-swizzled chunks)
            v16bf bfrag;
            {
                int colL = nSub * 16 + (lane & 15);
                int sw   = lane & 15;
                int c0   = (kb >> 3) + ((lane >> 4) << 1);
                const bf16_t* p0 = &wLds[(((g * 64 + colL) << 6) + ((c0    ) ^ sw)) * 8];
                const bf16_t* p1 = &wLds[(((g * 64 + colL) << 6) + ((c0 + 1) ^ sw)) * 8];
                ((v8bf*)&bfrag)[0] = *(const v8bf*)p0;
                ((v8bf*)&bfrag)[1] = *(const v8bf*)p1;
            }
#pragma unroll
            for (int i = 0; i < 4; ++i) {
                v16bf afrag;
                int row = i * 16 + (lane & 15);
                int k0  = (lane >> 4) * 16;
                const bf16_t* p = &hA[g * (64 * 40) + row * 40 + k0];
                ((v8bf*)&afrag)[0] = *(const v8bf*)p;
                ((v8bf*)&afrag)[1] = *(const v8bf*)(p + 8);
                acc[i] = __builtin_amdgcn_wmma_f32_16x16x32_bf16(
                    false, afrag, false, bfrag, (short)0, acc[i], false, false);
            }
        }
        __syncthreads();   // hA dead; gBuf (aliased) becomes live

        // Add gx (vector loads: contiguous along b) and deposit into LDS
        {
            int colL = nSub * 16 + (lane & 15);
            int colG = hTile * 64 + colL;
            long gbase = ((((long)d * T_STEPS + ti) * 4 + g) * 512 + colG) * 64;
#pragma unroll
            for (int i = 0; i < 4; ++i) {
                int rbase = i * 16 + ((lane & 16) >> 1);
                v8f gv = *(const v8f*)&gx[gbase + rbase];
#pragma unroll
                for (int r = 0; r < 8; ++r)
                    gBuf[g * 4096 + (rbase + r) * 64 + colL] = acc[i][r] + gv[r];
            }
        }
        __syncthreads();

        // Elementwise cell update + mask blend: 64x64 elems, 8 per thread
#pragma unroll
        for (int e = 0; e < 8; ++e) {
            int idx = e * 512 + tid;
            float iv = 1.0f / (1.0f + __expf(-gBuf[0 * 4096 + idx]));
            float fv = 1.0f / (1.0f + __expf(-gBuf[1 * 4096 + idx]));
            float gv = tanhf(gBuf[2 * 4096 + idx]);
            float ov = 1.0f / (1.0f + __expf(-gBuf[3 * 4096 + idx]));
            int b  = idx >> 6;
            int hc = idx & 63;
            int hg = hTile * 64 + hc;
            float cOld = cIn[b * HID + hg];
            float hOld = hIn[b * HID + hg];
            float cNew = fv * cOld + iv * gv;
            float hNew = ov * tanhf(cNew);
            float m  = mask[ti * BATCH + b];
            float hF = hNew * m + hOld * (1.0f - m);
            float cF = cNew * m + cOld * (1.0f - m);
            hOut[b * HID + hg] = hF;
            cOut[b * HID + hg] = cF;
            outp[((long)ti * BATCH + b) * (2 * HID) + d * HID + hg] = hF;
        }

        // Grid-wide barrier: all 16 blocks finish step s before step s+1 reads h
        __threadfence();
        __syncthreads();
        if (tid == 0) {
            atomicAdd(barrier_ctr, 1u);
            unsigned target = (unsigned)NBLK * (unsigned)(s + 1);
            while (__hip_atomic_load(barrier_ctr, __ATOMIC_RELAXED,
                                     __HIP_MEMORY_SCOPE_AGENT) < target) {
                __builtin_amdgcn_s_sleep(2);
            }
        }
        __syncthreads();
        __threadfence();
    }
}

// Copy final h/c states into the h_n / c_n output regions
__global__ void finalize_kernel(const float* __restrict__ h_fin,
                                const float* __restrict__ c_fin,
                                float* __restrict__ outp, int layer) {
    int i = blockIdx.x * blockDim.x + threadIdx.x;
    int n = 2 * BATCH * HID;
    if (i >= n) return;
    long base = (long)T_STEPS * BATCH * 2 * HID;
    outp[base + (long)layer * n + i] = h_fin[i];
    outp[base + 4L * BATCH * HID + (long)layer * n + i] = c_fin[i];
}

// ---------------------------------------------------------------------------
extern "C" void kernel_launch(void* const* d_in, const int* in_sizes, int n_in,
                              void* d_out, int out_size, void* d_ws, size_t ws_size,
                              hipStream_t stream) {
    (void)in_sizes; (void)n_in; (void)out_size; (void)ws_size;
    const float* x       = (const float*)d_in[0];
    const float* mask    = (const float*)d_in[1];
    const float* w_ih_l0 = (const float*)d_in[2];
    const float* w_hh_l0 = (const float*)d_in[3];
    const float* b_ih_l0 = (const float*)d_in[4];
    const float* b_hh_l0 = (const float*)d_in[5];
    const float* n_in_l0 = (const float*)d_in[6];
    const float* n_h_l0  = (const float*)d_in[7];
    const float* w_ih_l1 = (const float*)d_in[8];
    const float* w_hh_l1 = (const float*)d_in[9];
    const float* b_ih_l1 = (const float*)d_in[10];
    const float* b_hh_l1 = (const float*)d_in[11];
    const float* n_in_l1 = (const float*)d_in[12];
    const float* n_h_l1  = (const float*)d_in[13];
    float* out = (float*)d_out;

    char* ws = (char*)d_ws;
    size_t off = 0;
    auto alloc = [&](size_t bytes) -> void* {
        void* p = ws + off;
        off = (off + bytes + 255) & ~(size_t)255;
        return p;
    };
    bf16_t* wTih0 = (bf16_t*)alloc(8L * 512 * 512 * 2);
    bf16_t* wThh0 = (bf16_t*)alloc(8L * 512 * 512 * 2);
    bf16_t* wTih1 = (bf16_t*)alloc(8L * 1024 * 512 * 2);
    bf16_t* wThh1 = (bf16_t*)alloc(8L * 512 * 512 * 2);
    float*  gx    = (float*)alloc(2L * T_STEPS * 4 * HID * BATCH * 4);
    float*  out0  = (float*)alloc((long)T_STEPS * BATCH * 2 * HID * 4);
    float*  stateH[2], *stateC[2];
    for (int p = 0; p < 2; ++p) {
        stateH[p] = (float*)alloc(2L * BATCH * HID * 4);
        stateC[p] = (float*)alloc(2L * BATCH * HID * 4);
    }
    unsigned* barCtr = (unsigned*)alloc(256);

    {
        long t0 = 8L * 512 * 512;
        long t1 = 8L * 1024 * 512;
        wt_kernel<<<(int)((t0 + 255) / 256), 256, 0, stream>>>(w_ih_l0, wTih0, 512, 512, t0);
        wt_kernel<<<(int)((t0 + 255) / 256), 256, 0, stream>>>(w_hh_l0, wThh0, 512, 512, t0);
        wt_kernel<<<(int)((t1 + 255) / 256), 256, 0, stream>>>(w_ih_l1, wTih1, 1024, 512, t1);
        wt_kernel<<<(int)((t0 + 255) / 256), 256, 0, stream>>>(w_hh_l1, wThh1, 512, 512, t0);
    }

    const int stElems = 2 * BATCH * HID;
    auto run_layer = [&](const float* Xin, int K, const float* noise_in, const bf16_t* wTih,
                         const float* bih, const float* bhh, const float* noise_hid,
                         const bf16_t* wThh, float* layer_out, int layer) {
        for (int p = 0; p < 2; ++p) {
            zero_kernel<<<(stElems + 255) / 256, 256, 0, stream>>>(stateH[p], stElems);
            zero_kernel<<<(stElems + 255) / 256, 256, 0, stream>>>(stateC[p], stElems);
        }
        zero_kernel<<<1, 64, 0, stream>>>((float*)barCtr, 64);
        dim3 gg(128, 8, 2);
        gx_gemm_kernel<<<gg, 512, 0, stream>>>(Xin, noise_in, wTih, bih, bhh, gx, K);
        lstm_scan_kernel<<<dim3(8, 2), 512, 0, stream>>>(
            gx, mask, noise_hid, wThh,
            stateH[0], stateC[0], stateH[1], stateC[1],
            layer_out, barCtr);
        finalize_kernel<<<(stElems + 255) / 256, 256, 0, stream>>>(stateH[0], stateC[0], out, layer);
    };

    run_layer(x,    512,  n_in_l0, wTih0, b_ih_l0, b_hh_l0, n_h_l0, wThh0, out0, 0);
    run_layer(out0, 1024, n_in_l1, wTih1, b_ih_l1, b_hh_l1, n_h_l1, wThh1, out,  1);
}
